// EncoderBlock_13881334300931
// MI455X (gfx1250) — matrix operand order repaired and checked by reading, not compiled, not run on hardware
//
#include <hip/hip_runtime.h>
#include <hip/hip_bf16.h>

// ---------------------------------------------------------------------------
// CDNA5 / gfx1250 fused transformer encoder block.
// All GEMMs via v_wmma_f32_16x16x32_f16 (f16 operands, f32 accumulate),
// software-pipelined so WMMAs wait on loads issued one iteration earlier.
// ---------------------------------------------------------------------------

typedef _Float16 v16h __attribute__((ext_vector_type(16)));
typedef _Float16 v8h  __attribute__((ext_vector_type(8)));
typedef float    v8f  __attribute__((ext_vector_type(8)));

#define D_MODEL 512
#define FFN_DIM 2048
#define SEQ     4096
#define BATCH   4
#define MTOT    (BATCH * SEQ)

__device__ __forceinline__ v8f vzero8() {
  v8f r;
#pragma unroll
  for (int j = 0; j < 8; j++) r[j] = 0.0f;
  return r;
}

__device__ __forceinline__ v8f vsplat8(float x) {
  v8f r;
#pragma unroll
  for (int j = 0; j < 8; j++) r[j] = x;
  return r;
}

__device__ __forceinline__ v16h cat8(v8h lo, v8h hi) {
  return __builtin_shufflevector(lo, hi, 0, 1, 2, 3, 4, 5, 6, 7,
                                 8, 9, 10, 11, 12, 13, 14, 15);
}

// WMMA D = A(16x32 f16) * B(32x16 f16) + C(16x16 f32)
__device__ __forceinline__ v8f wmma_f16(v16h a, v16h b, v8f c) {
  return __builtin_amdgcn_wmma_f32_16x16x32_f16(
      /*neg_a=*/false, a, /*neg_b=*/false, b,
      /*c_mod=*/(short)0, c, /*reuse_a=*/false, /*reuse_b=*/false);
}

// A fragment: row-major [16 rows x K-slice], base points at (m0, kb), pitch ld.
__device__ __forceinline__ v16h load_afrag(const _Float16* base, int ld, int lane) {
  const int m   = lane & 15;
  const int kb8 = (lane & 16) ? 8 : 0;
  const _Float16* p = base + (size_t)m * ld + kb8;
  v8h lo = *(const v8h*)(p);
  v8h hi = *(const v8h*)(p + 16);
  return cat8(lo, hi);
}

// B fragment: Bt stored row-major [N][K]; base points at (n0, kb), pitch ld.
__device__ __forceinline__ v16h load_bfrag(const _Float16* base, int ld, int lane) {
  const int n    = lane & 15;
  const int koff = (lane & 16) ? 8 : 0;
  const _Float16* p = base + (size_t)n * ld + koff;
  return cat8(*(const v8h*)(p), *(const v8h*)(p + 16));
}

// ---------------------------------------------------------------------------
// Prep kernels
// ---------------------------------------------------------------------------
__global__ void cast_f32_f16(const float* __restrict__ in,
                             _Float16* __restrict__ out, int n) {
  int i = blockIdx.x * 256 + threadIdx.x;
  if (i < n) out[i] = (_Float16)in[i];
}

// in: [K][N] f32 row-major -> out: [N][K] f16 row-major
__global__ void transpose_cast(const float* __restrict__ in,
                               _Float16* __restrict__ out, int K, int N) {
  int i = blockIdx.x * 256 + threadIdx.x;
  if (i < K * N) {
    int k = i / N, n = i % N;
    out[(size_t)n * K + k] = (_Float16)in[i];
  }
}

// ---------------------------------------------------------------------------
// QKV projection: one wave per 16x64 tile of Q, K or V (blockIdx.z).
// Double-buffered K loop: loads for kb+32 issued before WMMAs for kb.
// ---------------------------------------------------------------------------
__global__ void qkv_kernel(const _Float16* __restrict__ xb,
                           const _Float16* __restrict__ Wqt,
                           const _Float16* __restrict__ Wkt,
                           const _Float16* __restrict__ Wvt,
                           const float* __restrict__ bq,
                           const float* __restrict__ bk,
                           const float* __restrict__ bv,
                           _Float16* __restrict__ Qb,
                           _Float16* __restrict__ Kb,
                           _Float16* __restrict__ Vt) {
  const int lane = threadIdx.x & 31;
  const int row0 = blockIdx.x * 16;
  const int n0   = blockIdx.y * 64;
  const int proj = blockIdx.z;

  const _Float16* Wt   = (proj == 0) ? Wqt : (proj == 1) ? Wkt : Wvt;
  const float*    bias = (proj == 0) ? bq : (proj == 1) ? bk : bv;

  v8f acc[4];
#pragma unroll
  for (int t = 0; t < 4; t++)
    acc[t] = vsplat8(bias[n0 + t * 16 + (lane & 15)]);

  const _Float16* abase = xb + (size_t)row0 * D_MODEL;

  // prologue
  v16h a_cur = load_afrag(abase, D_MODEL, lane);
  v16h b_cur[4];
#pragma unroll
  for (int t = 0; t < 4; t++)
    b_cur[t] = load_bfrag(Wt + (size_t)(n0 + t * 16) * D_MODEL, D_MODEL, lane);

  for (int kb = 0; kb < D_MODEL - 32; kb += 32) {
    v16h a_nxt = load_afrag(abase + kb + 32, D_MODEL, lane);
    v16h b_nxt[4];
#pragma unroll
    for (int t = 0; t < 4; t++)
      b_nxt[t] = load_bfrag(Wt + (size_t)(n0 + t * 16) * D_MODEL + kb + 32,
                            D_MODEL, lane);
#pragma unroll
    for (int t = 0; t < 4; t++)
      acc[t] = wmma_f16(a_cur, b_cur[t], acc[t]);
    a_cur = a_nxt;
#pragma unroll
    for (int t = 0; t < 4; t++) b_cur[t] = b_nxt[t];
  }
  // epilogue
#pragma unroll
  for (int t = 0; t < 4; t++)
    acc[t] = wmma_f16(a_cur, b_cur[t], acc[t]);

  const float qscale = 0.04419417382415922f;  // 1/sqrt(512)
  const int rr = (lane & 16) ? 8 : 0;
#pragma unroll
  for (int t = 0; t < 4; t++) {
#pragma unroll
    for (int j = 0; j < 8; j++) {
      const int row = row0 + j + rr;
      const int col = n0 + t * 16 + (lane & 15);
      const float v = acc[t][j];
      if (proj == 0) {
        Qb[(size_t)row * D_MODEL + col] = (_Float16)(v * qscale);
      } else if (proj == 1) {
        Kb[(size_t)row * D_MODEL + col] = (_Float16)v;
      } else {
        const int bt = row >> 12;
        const int s  = row & 4095;
        Vt[((size_t)bt * D_MODEL + col) * SEQ + s] = (_Float16)v;
      }
    }
  }
}

// ---------------------------------------------------------------------------
// Attention: one workgroup (4 waves) per 16-query block. Each wave holds all
// Q fragments in registers, computes the full 16x32 score tile, exp + row-sum,
// reshapes P through a per-wave LDS tile, accumulates its 128-col O slice.
// V fragments are loaded *before* the softmax math so their latency hides
// under exp/shfl/ds work. K/V streams are prefetched one chunk ahead.
// ---------------------------------------------------------------------------
__global__ void attn_kernel(const _Float16* __restrict__ Qb,
                            const _Float16* __restrict__ Kb,
                            const _Float16* __restrict__ Vt,
                            float* __restrict__ att) {
  __shared__ __align__(32) _Float16 pbuf[4][16][40];

  const int lane  = threadIdx.x & 31;
  const int wave  = threadIdx.x >> 5;
  const int row0  = blockIdx.x * 16;
  const int batch = row0 >> 12;
  const int dbase = wave * 128;
  const int rr    = (lane & 16) ? 8 : 0;

  // All Q A-fragments (16 rows x full d=512) in registers.
  v16h qf[16];
#pragma unroll
  for (int kd = 0; kd < 16; kd++)
    qf[kd] = load_afrag(Qb + (size_t)row0 * D_MODEL + kd * 32, D_MODEL, lane);

  v8f acc[8];
#pragma unroll
  for (int t = 0; t < 8; t++) acc[t] = vzero8();
  v8f lsum = vzero8();

  for (int k0 = 0; k0 < SEQ; k0 += 32) {
    const _Float16* kbp = Kb + ((size_t)batch * SEQ + k0) * D_MODEL;
    const _Float16* vbp = Vt + ((size_t)batch * D_MODEL + dbase) * SEQ + k0;

    // speculative prefetch of next chunk's K rows / V columns
    __builtin_prefetch(kbp + (size_t)32 * D_MODEL + (size_t)lane * 64, 0, 0);
    __builtin_prefetch(vbp + 32 + (size_t)(lane & 15) * SEQ, 0, 0);

    // ---- scores tile 16(q) x 32(k), double-buffered over d ----
    v8f s0 = vzero8(), s1 = vzero8();
    v16h k0c = load_bfrag(kbp, D_MODEL, lane);
    v16h k1c = load_bfrag(kbp + (size_t)16 * D_MODEL, D_MODEL, lane);
#pragma unroll
    for (int kd = 0; kd < 16; kd++) {
      v16h k0n = k0c, k1n = k1c;
      if (kd + 1 < 16) {  // statically resolved under full unroll
        k0n = load_bfrag(kbp + (kd + 1) * 32, D_MODEL, lane);
        k1n = load_bfrag(kbp + (size_t)16 * D_MODEL + (kd + 1) * 32, D_MODEL, lane);
      }
      s0 = wmma_f16(qf[kd], k0c, s0);
      s1 = wmma_f16(qf[kd], k1c, s1);
      k0c = k0n;
      k1c = k1n;
    }

    // ---- issue V fragment loads now; they are independent of P ----
    v16h bf[8];
#pragma unroll
    for (int dt = 0; dt < 8; dt++)
      bf[dt] = load_bfrag(vbp + (size_t)dt * 16 * SEQ, SEQ, lane);

    // ---- exp, per-row sums, stash P tile in LDS (overlaps V loads) ----
#pragma unroll
    for (int j = 0; j < 8; j++) {
      float p0 = __expf(s0[j]);
      float p1 = __expf(s1[j]);
      float rsum = p0 + p1;
      rsum += __shfl_xor(rsum, 1);
      rsum += __shfl_xor(rsum, 2);
      rsum += __shfl_xor(rsum, 4);
      rsum += __shfl_xor(rsum, 8);
      lsum[j] += rsum;
      pbuf[wave][j + rr][lane & 15]        = (_Float16)p0;
      pbuf[wave][j + rr][16 + (lane & 15)] = (_Float16)p1;
    }

    // ---- reload P as A fragment (same-wave ds ops are in-order) ----
    {
      const int m   = lane & 15;
      const int kb8 = (lane & 16) ? 8 : 0;
      v8h lo = *(const v8h*)&pbuf[wave][m][kb8];
      v8h hi = *(const v8h*)&pbuf[wave][m][16 + kb8];
      v16h pf = cat8(lo, hi);

#pragma unroll
      for (int dt = 0; dt < 8; dt++)
        acc[dt] = wmma_f16(pf, bf[dt], acc[dt]);
    }
  }

  // normalize and write out (f32)
#pragma unroll
  for (int dt = 0; dt < 8; dt++) {
#pragma unroll
    for (int j = 0; j < 8; j++) {
      const int row = row0 + j + rr;
      const int col = dbase + dt * 16 + (lane & 15);
      att[(size_t)row * D_MODEL + col] = acc[dt][j] / lsum[j];
    }
  }
}

// ---------------------------------------------------------------------------
// Residual add + LayerNorm: one wave per row, shfl_xor reductions.
// ---------------------------------------------------------------------------
__global__ void addln_kernel(const float* __restrict__ a,
                             const float* __restrict__ r,
                             const float* __restrict__ g,
                             const float* __restrict__ bb,
                             float* __restrict__ out,
                             _Float16* __restrict__ outh) {
  const int lane = threadIdx.x & 31;
  const int wave = threadIdx.x >> 5;
  const int row  = blockIdx.x * 8 + wave;

  const float* pa = a + (size_t)row * D_MODEL;
  const float* pr = r + (size_t)row * D_MODEL;

  float v[16];
  float s = 0.0f, sq = 0.0f;
#pragma unroll
  for (int i = 0; i < 16; i++) {
    float t = pa[lane + i * 32] + pr[lane + i * 32];
    v[i] = t;
    s += t;
    sq += t * t;
  }
#pragma unroll
  for (int m = 1; m < 32; m <<= 1) {
    s  += __shfl_xor(s, m);
    sq += __shfl_xor(sq, m);
  }
  const float mu  = s * (1.0f / D_MODEL);
  const float var = sq * (1.0f / D_MODEL) - mu * mu;
  const float rs  = rsqrtf(var + 1e-5f);

#pragma unroll
  for (int i = 0; i < 16; i++) {
    const int c = lane + i * 32;
    const float o = (v[i] - mu) * rs * g[c] + bb[c];
    out[(size_t)row * D_MODEL + c] = o;
    if (outh) outh[(size_t)row * D_MODEL + c] = (_Float16)o;
  }
}

// ---------------------------------------------------------------------------
// Fused FFN: per 16-row block, h = relu(x1*W1+b1) built in LDS (16x2048 f16),
// then y = h*W2+b2 consumed straight from LDS. 4 waves split the N dims.
// Tile groups of 4 with full double buffering (fits ~130 VGPRs).
// ---------------------------------------------------------------------------
__global__ void ffn_kernel(const _Float16* __restrict__ x1h,
                           const _Float16* __restrict__ W1t,
                           const float* __restrict__ b1f,
                           const _Float16* __restrict__ W2t,
                           const float* __restrict__ b2f,
                           float* __restrict__ y) {
  extern __shared__ _Float16 hbuf[];   // [16][HP] f16
  const int HP = FFN_DIM + 16;         // 2064: 16B-aligned rows, de-rotated banks

  const int lane = threadIdx.x & 31;
  const int wave = threadIdx.x >> 5;
  const int row0 = blockIdx.x * 16;
  const int rr   = (lane & 16) ? 8 : 0;
  const _Float16* abase = x1h + (size_t)row0 * D_MODEL;

  // ---- FFN1: this wave produces columns [wave*512, wave*512+512) of h ----
  for (int g = 0; g < 8; g++) {
    const int nb = wave * 512 + g * 64;
    v8f acc[4];
#pragma unroll
    for (int t = 0; t < 4; t++)
      acc[t] = vsplat8(b1f[nb + t * 16 + (lane & 15)]);

    v16h a_cur = load_afrag(abase, D_MODEL, lane);
    v16h b_cur[4];
#pragma unroll
    for (int t = 0; t < 4; t++)
      b_cur[t] = load_bfrag(W1t + (size_t)(nb + t * 16) * D_MODEL, D_MODEL, lane);

    for (int kb = 0; kb < D_MODEL - 32; kb += 32) {
      v16h a_nxt = load_afrag(abase + kb + 32, D_MODEL, lane);
      v16h b_nxt[4];
#pragma unroll
      for (int t = 0; t < 4; t++)
        b_nxt[t] = load_bfrag(W1t + (size_t)(nb + t * 16) * D_MODEL + kb + 32,
                              D_MODEL, lane);
#pragma unroll
      for (int t = 0; t < 4; t++)
        acc[t] = wmma_f16(a_cur, b_cur[t], acc[t]);
      a_cur = a_nxt;
#pragma unroll
      for (int t = 0; t < 4; t++) b_cur[t] = b_nxt[t];
    }
#pragma unroll
    for (int t = 0; t < 4; t++)
      acc[t] = wmma_f16(a_cur, b_cur[t], acc[t]);

#pragma unroll
    for (int t = 0; t < 4; t++) {
#pragma unroll
      for (int j = 0; j < 8; j++) {
        float v = acc[t][j];
        v = v > 0.0f ? v : 0.0f;  // ReLU
        hbuf[(size_t)(j + rr) * HP + nb + t * 16 + (lane & 15)] = (_Float16)v;
      }
    }
  }

  __syncthreads();

  // ---- FFN2: this wave produces columns [wave*128, wave*128+128) of y ----
  for (int g = 0; g < 2; g++) {
    const int nb2 = wave * 128 + g * 64;
    v8f acc2[4];
#pragma unroll
    for (int t = 0; t < 4; t++)
      acc2[t] = vsplat8(b2f[nb2 + t * 16 + (lane & 15)]);

    const int m   = lane & 15;
    const int kb8 = (lane & 16) ? 8 : 0;
    const _Float16* hrow = hbuf + (size_t)m * HP + kb8;

    v16h a_cur = cat8(*(const v8h*)(hrow), *(const v8h*)(hrow + 16));
    v16h b_cur[4];
#pragma unroll
    for (int t = 0; t < 4; t++)
      b_cur[t] = load_bfrag(W2t + (size_t)(nb2 + t * 16) * FFN_DIM, FFN_DIM, lane);

    for (int kb = 0; kb < FFN_DIM - 32; kb += 32) {
      v16h a_nxt = cat8(*(const v8h*)(hrow + kb + 32),
                        *(const v8h*)(hrow + kb + 48));
      v16h b_nxt[4];
#pragma unroll
      for (int t = 0; t < 4; t++)
        b_nxt[t] = load_bfrag(W2t + (size_t)(nb2 + t * 16) * FFN_DIM + kb + 32,
                              FFN_DIM, lane);
#pragma unroll
      for (int t = 0; t < 4; t++)
        acc2[t] = wmma_f16(a_cur, b_cur[t], acc2[t]);
      a_cur = a_nxt;
#pragma unroll
      for (int t = 0; t < 4; t++) b_cur[t] = b_nxt[t];
    }
#pragma unroll
    for (int t = 0; t < 4; t++)
      acc2[t] = wmma_f16(a_cur, b_cur[t], acc2[t]);

#pragma unroll
    for (int t = 0; t < 4; t++) {
#pragma unroll
      for (int j = 0; j < 8; j++) {
        const int row = row0 + j + rr;
        const int col = nb2 + t * 16 + (lane & 15);
        y[(size_t)row * D_MODEL + col] = acc2[t][j];
      }
    }
  }
}

// ---------------------------------------------------------------------------
// Host launcher
// ---------------------------------------------------------------------------
extern "C" void kernel_launch(void* const* d_in, const int* in_sizes, int n_in,
                              void* d_out, int out_size, void* d_ws, size_t ws_size,
                              hipStream_t stream) {
  const float* x   = (const float*)d_in[0];
  const float* Wq  = (const float*)d_in[1];
  const float* bq  = (const float*)d_in[2];
  const float* Wk  = (const float*)d_in[3];
  const float* bk  = (const float*)d_in[4];
  const float* Wv  = (const float*)d_in[5];
  const float* bv  = (const float*)d_in[6];
  const float* g1  = (const float*)d_in[7];
  const float* b1  = (const float*)d_in[8];
  const float* g2  = (const float*)d_in[9];
  const float* b2  = (const float*)d_in[10];
  const float* W1  = (const float*)d_in[11];
  const float* bf1 = (const float*)d_in[12];
  const float* W2  = (const float*)d_in[13];
  const float* bf2 = (const float*)d_in[14];
  float* out = (float*)d_out;

  char* ws = (char*)d_ws;
  size_t off = 0;
  auto alloc = [&](size_t bytes) -> char* {
    char* p = ws + off;
    off += (bytes + 255) & ~(size_t)255;
    return p;
  };

  const size_t MD = (size_t)MTOT * D_MODEL;
  _Float16* xb  = (_Float16*)alloc(MD * 2);
  _Float16* Wqt = (_Float16*)alloc((size_t)D_MODEL * D_MODEL * 2);
  _Float16* Wkt = (_Float16*)alloc((size_t)D_MODEL * D_MODEL * 2);
  _Float16* Wvt = (_Float16*)alloc((size_t)D_MODEL * D_MODEL * 2);
  _Float16* W1t = (_Float16*)alloc((size_t)FFN_DIM * D_MODEL * 2);
  _Float16* W2t = (_Float16*)alloc((size_t)D_MODEL * FFN_DIM * 2);
  _Float16* Qb  = (_Float16*)alloc(MD * 2);
  _Float16* Kb  = (_Float16*)alloc(MD * 2);
  _Float16* Vt  = (_Float16*)alloc(MD * 2);
  float*    att = (float*)alloc(MD * 4);
  float*    x1  = (float*)alloc(MD * 4);
  _Float16* x1h = (_Float16*)alloc(MD * 2);
  float*    yb  = (float*)alloc(MD * 4);

  // --- prep: casts / transposes ---
  {
    const int nx = MTOT * D_MODEL;
    cast_f32_f16<<<(nx + 255) / 256, 256, 0, stream>>>(x, xb, nx);
    const int ndd = D_MODEL * D_MODEL;
    transpose_cast<<<(ndd + 255) / 256, 256, 0, stream>>>(Wq, Wqt, D_MODEL, D_MODEL);
    transpose_cast<<<(ndd + 255) / 256, 256, 0, stream>>>(Wk, Wkt, D_MODEL, D_MODEL);
    transpose_cast<<<(ndd + 255) / 256, 256, 0, stream>>>(Wv, Wvt, D_MODEL, D_MODEL);
    const int ndf = D_MODEL * FFN_DIM;
    transpose_cast<<<(ndf + 255) / 256, 256, 0, stream>>>(W1, W1t, D_MODEL, FFN_DIM);
    transpose_cast<<<(ndf + 255) / 256, 256, 0, stream>>>(W2, W2t, FFN_DIM, D_MODEL);
  }

  // --- QKV projections (WMMA) ---
  qkv_kernel<<<dim3(MTOT / 16, D_MODEL / 64, 3), 32, 0, stream>>>(
      xb, Wqt, Wkt, Wvt, bq, bk, bv, Qb, Kb, Vt);

  // --- attention (WMMA scores + WMMA P*V, streaming softmax) ---
  attn_kernel<<<MTOT / 16, 128, 0, stream>>>(Qb, Kb, Vt, att);

  // --- residual + LN1 (also emits f16 copy for FFN A-fragments) ---
  addln_kernel<<<MTOT / 8, 256, 0, stream>>>(x, att, g1, b1, x1, x1h);

  // --- fused FFN (WMMA, h lives in LDS) ---
  const size_t ffn_lds = (size_t)16 * (FFN_DIM + 16) * sizeof(_Float16);  // 66048 B
  ffn_kernel<<<MTOT / 16, 128, ffn_lds, stream>>>(x1h, W1t, bf1, W2t, bf2, yb);

  // --- residual + LN2 -> final output ---
  addln_kernel<<<MTOT / 8, 256, 0, stream>>>(x1, yb, g2, b2, out, nullptr);
}
